// SparseMoE_68728066670792
// MI455X (gfx1250) — matrix operand order repaired and checked by reading, not compiled, not run on hardware
//
#include <hip/hip_runtime.h>
#include <math.h>

#define T    2048
#define H    512
#define E    32
#define K    4
#define FFN  2048

#define MT          32    // rows (gathered tokens) per block
#define XS_PITCH    520   // 512 + 8 bf16 pad (row stride 1040B, 16B-aligned)
#define HS_PITCH    2056  // 2048 + 8 bf16 pad (row stride 4112B, 16B-aligned)
#define TILES_PER_E 64    // ceil(2048 / 32)

typedef __bf16 bf16_t;
typedef __attribute__((ext_vector_type(16))) __bf16 v16bf;
typedef __attribute__((ext_vector_type(8)))  __bf16 v8bf;
typedef __attribute__((ext_vector_type(8)))  float  v8f;

__device__ __forceinline__ v8f wmma_bf16(v16bf a, v16bf b, v8f c) {
  return __builtin_amdgcn_wmma_f32_16x16x32_bf16(false, a, false, b,
                                                 (short)0, c, false, false);
}

// Per-lane A fragment for 16x16x32 bf16 WMMA from a row-major bf16 LDS tile.
// Lane holds row (lane&15); its 16 K-values are two contiguous 16B chunks.
__device__ __forceinline__ v16bf load_a_frag(const bf16_t* base, int pitch,
                                             int lm, int hi, int kb) {
  v8bf c0 = *(const v8bf*)&base[lm * pitch + kb + 16 * hi];
  v8bf c1 = *(const v8bf*)&base[lm * pitch + kb + 16 * hi + 8];
  return __builtin_shufflevector(c0, c1, 0, 1, 2, 3, 4, 5, 6, 7,
                                 8, 9, 10, 11, 12, 13, 14, 15);
}

// ---------------------------------------------------------------------------
// Kernel 1: router. One thread per token: logits, top-4 + softmax, full probs.
// ---------------------------------------------------------------------------
__global__ void moe_router(const float* __restrict__ x, const float* __restrict__ gw,
                           int* __restrict__ topi, float* __restrict__ topw,
                           float* __restrict__ probs) {
  int t = blockIdx.x * blockDim.x + threadIdx.x;
  if (t >= T) return;
  float acc[E];
#pragma unroll
  for (int e = 0; e < E; ++e) acc[e] = 0.f;
  const float* xr = x + (size_t)t * H;
  for (int h = 0; h < H; ++h) {
    float xv = xr[h];
    const float* gr = gw + (size_t)h * E;
#pragma unroll
    for (int e = 0; e < E; ++e) acc[e] += xv * gr[e];
  }
  // top-4 (strict > so ties pick the lowest index, like jax.lax.top_k)
  unsigned chosen = 0u;
  int bi[K]; float bv[K];
#pragma unroll
  for (int j = 0; j < K; ++j) {
    float best = -INFINITY; int besti = 0;
#pragma unroll
    for (int e = 0; e < E; ++e) {
      bool taken = (chosen >> e) & 1u;
      if (!taken && acc[e] > best) { best = acc[e]; besti = e; }
    }
    chosen |= 1u << besti; bi[j] = besti; bv[j] = best;
  }
  float m = bv[0];                 // top-1 is also the global max
  float s = 0.f; float w[K];
#pragma unroll
  for (int j = 0; j < K; ++j) { w[j] = __expf(bv[j] - m); s += w[j]; }
  float inv = 1.f / s;
#pragma unroll
  for (int j = 0; j < K; ++j) { topi[t*K + j] = bi[j]; topw[t*K + j] = w[j] * inv; }
  // full softmax probs (for aux loss), streamed in two passes
  float ps = 0.f;
#pragma unroll
  for (int e = 0; e < E; ++e) ps += __expf(acc[e] - m);
  float pinv = 1.f / ps;
#pragma unroll
  for (int e = 0; e < E; ++e) probs[(size_t)t*E + e] = __expf(acc[e] - m) * pinv;
}

// ---------------------------------------------------------------------------
// Kernel 2: deterministic per-expert gather lists (one wave, thread = expert).
// ltok stores packed index (t*4 + slot) == contrib row id.
// ---------------------------------------------------------------------------
__global__ void moe_build_lists(const int* __restrict__ topi, const float* __restrict__ topw,
                                int* __restrict__ counts, int* __restrict__ ltok,
                                float* __restrict__ lwt) {
  int e = threadIdx.x;
  if (e >= E) return;
  int cnt = 0;
  for (int t = 0; t < T; ++t) {
#pragma unroll
    for (int k = 0; k < K; ++k) {
      int idx = t*K + k;
      if (topi[idx] == e) { ltok[e*T + cnt] = idx; lwt[e*T + cnt] = topw[idx]; ++cnt; }
    }
  }
  counts[e] = cnt;
}

// ---------------------------------------------------------------------------
// Kernel 3: aux loss. S_e = sum_t probs[t][e]; aux = E * sum (S/T)*(S/sumS).
// ---------------------------------------------------------------------------
__global__ void moe_aux(const float* __restrict__ probs, float* __restrict__ aux_out) {
  __shared__ float S[E];
  int e = threadIdx.x;
  if (e < E) {
    float s = 0.f;
    for (int t = 0; t < T; ++t) s += probs[(size_t)t*E + e];
    S[e] = s;
  }
  __syncthreads();
  if (threadIdx.x == 0) {
    float tot = 0.f;
    for (int i = 0; i < E; ++i) tot += S[i];
    float a = 0.f;
    for (int i = 0; i < E; ++i) a += (S[i] / (float)T) * (S[i] / tot);
    aux_out[0] = (float)E * a;
  }
}

// ---------------------------------------------------------------------------
// Kernel 4: expert FFN. Block = 32 gathered rows of one expert, 256 thr/8 waves.
// LDS: xs 32x512 bf16 (32KB), hs 32x2048 bf16 (128KB): h never leaves the WGP.
// Two 16-row WMMA tiles share every B fragment (2x FLOPs per weight byte).
// GEMM1/3: each wave owns 256 FFN cols; GEMM2: each wave owns 64 H cols.
// ---------------------------------------------------------------------------
__global__ __launch_bounds__(256) void moe_expert_ffn(
    const float* __restrict__ x,  const float* __restrict__ w1,
    const float* __restrict__ w2, const float* __restrict__ w3,
    const int* __restrict__ counts, const int* __restrict__ ltok,
    const float* __restrict__ lwt, float* __restrict__ contrib) {
  extern __shared__ char smem_raw[];
  int*    ts  = (int*)smem_raw;                               // 32 packed (t*4+slot)
  float*  cws = (float*)(smem_raw + 128);                     // 32 routing weights
  bf16_t* xs  = (bf16_t*)(smem_raw + 256);                    // 32 x XS_PITCH
  bf16_t* hs  = (bf16_t*)(smem_raw + 256 + MT*XS_PITCH*2);    // 32 x HS_PITCH

  int e    = blockIdx.x >> 6;
  int tile = blockIdx.x & (TILES_PER_E - 1);
  int Ne   = counts[e];
  int row0 = tile * MT;
  if (row0 >= Ne) return;                 // uniform per block
  int nvalid = Ne - row0; if (nvalid > MT) nvalid = MT;

  int tid = threadIdx.x;
  if (tid < MT) {
    if (tid < nvalid) { ts[tid] = ltok[e*T + row0 + tid]; cws[tid] = lwt[e*T + row0 + tid]; }
    else              { ts[tid] = 0;                      cws[tid] = 0.f; }
  }
  __syncthreads();

  // gather X rows (token = ts>>2) into LDS as bf16; pad rows are zero
  for (int idx = tid; idx < MT*H; idx += 256) {
    int r = idx >> 9, c = idx & (H - 1);
    float v = (r < nvalid) ? x[(size_t)(ts[r] >> 2) * H + c] : 0.f;
    xs[r*XS_PITCH + c] = (bf16_t)v;
  }
  __syncthreads();

  int wave = tid >> 5, lane = tid & 31;
  int hi = lane >> 4, lm = lane & 15;
  const float* w1e = w1 + (size_t)e * H * FFN;
  const float* w3e = w3 + (size_t)e * H * FFN;
  const float* w2e = w2 + (size_t)e * FFN * H;

  // ---- GEMM1 + GEMM3 + silu*mul fused: h[32 x 2048] ----
  int cb = wave * 256;
  for (int nt = 0; nt < 16; ++nt) {
    int ncol = cb + nt * 16;
    v8f acc1_0 = {}, acc1_1 = {}, acc3_0 = {}, acc3_1 = {};
    for (int kb = 0; kb < H; kb += 32) {
      if (kb + 32 < H) {
        __builtin_prefetch(w1e + (size_t)(kb + 32) * FFN + ncol, 0, 1);
        __builtin_prefetch(w3e + (size_t)(kb + 32) * FFN + ncol, 0, 1);
      }
      v16bf a0 = load_a_frag(xs, XS_PITCH, lm,      hi, kb);  // rows 0..15
      v16bf a1 = load_a_frag(xs + 16*XS_PITCH, XS_PITCH, lm, hi, kb); // rows 16..31
      v16bf b1, b3;
#pragma unroll
      for (int v = 0; v < 8; ++v) {   // B 32x16 bf16 layout: N=lm, K=2v+16*hi
        int kk = kb + 2*v + 16*hi;
        size_t off = (size_t)kk * FFN + ncol + lm;
        b1[2*v]     = (bf16_t)w1e[off];
        b1[2*v + 1] = (bf16_t)w1e[off + FFN];
        b3[2*v]     = (bf16_t)w3e[off];
        b3[2*v + 1] = (bf16_t)w3e[off + FFN];
      }
      acc1_0 = wmma_bf16(a0, b1, acc1_0);
      acc1_1 = wmma_bf16(a1, b1, acc1_1);
      acc3_0 = wmma_bf16(a0, b3, acc3_0);
      acc3_1 = wmma_bf16(a1, b3, acc3_1);
    }
#pragma unroll
    for (int r = 0; r < 8; ++r) {     // C layout: M = r + 8*hi, N = lm
      int m = r + 8*hi;
      float g0 = acc1_0[r];
      float h0 = (g0 / (1.f + __expf(-g0))) * acc3_0[r];
      hs[m*HS_PITCH + ncol + lm] = (bf16_t)h0;
      float g1 = acc1_1[r];
      float h1 = (g1 / (1.f + __expf(-g1))) * acc3_1[r];
      hs[(m + 16)*HS_PITCH + ncol + lm] = (bf16_t)h1;
    }
  }
  __syncthreads();

  // ---- GEMM2: out[32 x 512] = h @ w2[e], scaled by routing weight ----
  int cb2 = wave * 64;
  for (int nt = 0; nt < 4; ++nt) {
    int ncol = cb2 + nt * 16;
    v8f acc_0 = {}, acc_1 = {};
    for (int kb = 0; kb < FFN; kb += 32) {
      if (kb + 32 < FFN) {
        __builtin_prefetch(w2e + (size_t)(kb + 32) * H + ncol, 0, 1);
      }
      v16bf a0 = load_a_frag(hs, HS_PITCH, lm, hi, kb);
      v16bf a1 = load_a_frag(hs + 16*HS_PITCH, HS_PITCH, lm, hi, kb);
      v16bf b;
#pragma unroll
      for (int v = 0; v < 8; ++v) {
        int kk = kb + 2*v + 16*hi;
        size_t off = (size_t)kk * H + ncol + lm;
        b[2*v]     = (bf16_t)w2e[off];
        b[2*v + 1] = (bf16_t)w2e[off + H];
      }
      acc_0 = wmma_bf16(a0, b, acc_0);
      acc_1 = wmma_bf16(a1, b, acc_1);
    }
#pragma unroll
    for (int r = 0; r < 8; ++r) {
      int m = r + 8*hi;
      if (m < nvalid) {   // each (token,slot) written exactly once -> deterministic
        contrib[(size_t)ts[m] * H + ncol + lm] = acc_0[r] * cws[m];
      }
      int m1 = m + 16;
      if (m1 < nvalid) {
        contrib[(size_t)ts[m1] * H + ncol + lm] = acc_1[r] * cws[m1];
      }
    }
  }
}

// ---------------------------------------------------------------------------
// Kernel 5: combine the 4 slot contributions per token in fixed order.
// ---------------------------------------------------------------------------
__global__ void moe_combine(const float* __restrict__ contrib, float* __restrict__ out) {
  int i = blockIdx.x * blockDim.x + threadIdx.x;
  if (i >= T*H) return;
  int t = i >> 9, c = i & (H - 1);
  size_t base = (size_t)(t * K) * H + c;
  out[i] = contrib[base] + contrib[base + H] + contrib[base + 2*H] + contrib[base + 3*H];
}

// ---------------------------------------------------------------------------
extern "C" void kernel_launch(void* const* d_in, const int* in_sizes, int n_in,
                              void* d_out, int out_size, void* d_ws, size_t ws_size,
                              hipStream_t stream) {
  const float* x  = (const float*)d_in[0];
  const float* gw = (const float*)d_in[1];
  const float* w1 = (const float*)d_in[2];
  const float* w2 = (const float*)d_in[3];
  const float* w3 = (const float*)d_in[4];
  float* out = (float*)d_out;

  char* ws = (char*)d_ws;
  size_t off = 0;
  auto alloc = [&](size_t bytes) {
    off = (off + 255) & ~(size_t)255;
    char* p = ws + off; off += bytes; return p;
  };
  int*   topi    = (int*)  alloc((size_t)T*K*sizeof(int));
  float* topw    = (float*)alloc((size_t)T*K*sizeof(float));
  float* probs   = (float*)alloc((size_t)T*E*sizeof(float));
  int*   counts  = (int*)  alloc((size_t)E*sizeof(int));
  int*   ltok    = (int*)  alloc((size_t)E*T*sizeof(int));
  float* lwt     = (float*)alloc((size_t)E*T*sizeof(float));
  float* contrib = (float*)alloc((size_t)T*K*H*sizeof(float));
  (void)ws_size; (void)in_sizes; (void)n_in; (void)out_size;

  moe_router<<<(T + 255)/256, 256, 0, stream>>>(x, gw, topi, topw, probs);
  moe_build_lists<<<1, 32, 0, stream>>>(topi, topw, counts, ltok, lwt);
  moe_aux<<<1, 32, 0, stream>>>(probs, out + (size_t)T*H);

  size_t smem = 256 + (size_t)MT*XS_PITCH*2 + (size_t)MT*HS_PITCH*2;  // ~161 KB of 320 KB
  moe_expert_ffn<<<E*TILES_PER_E, 256, smem, stream>>>(x, w1, w2, w3,
                                                       counts, ltok, lwt, contrib);
  moe_combine<<<(T*H + 255)/256, 256, 0, stream>>>(contrib, out);
}
